// BCA_41369124995876
// MI455X (gfx1250) — matrix operand order repaired and checked by reading, not compiled, not run on hardware
//
#include <hip/hip_runtime.h>
#include <hip/hip_bf16.h>
#include <math.h>

// ---------------------------------------------------------------------------
// Types
// ---------------------------------------------------------------------------
typedef __bf16 bf16;
typedef __attribute__((ext_vector_type(16))) __bf16 v16bf;
typedef __attribute__((ext_vector_type(8)))  __bf16 v8bf;
typedef __attribute__((ext_vector_type(8)))  float  v8f;

#define C_DIM 1024
#define NUM_HEAD 8
#define DH 128

__device__ inline v8f wmma_bf16(v16bf a, v16bf b, v8f c) {
  return __builtin_amdgcn_wmma_f32_16x16x32_bf16(false, a, false, b, (short)0, c,
                                                 false, false);
}

__device__ inline v16bf cat8(v8bf lo, v8bf hi) {
  return __builtin_shufflevector(lo, hi, 0, 1, 2, 3, 4, 5, 6, 7, 8, 9, 10, 11,
                                 12, 13, 14, 15);
}

__device__ inline float hmax16(float v) {
  for (int m = 8; m >= 1; m >>= 1) v = fmaxf(v, __shfl_xor(v, m, 16));
  return v;
}
__device__ inline float hsum16(float v) {
  for (int m = 8; m >= 1; m >>= 1) v += __shfl_xor(v, m, 16);
  return v;
}
__device__ inline float wsum32(float v) {
  for (int m = 16; m >= 1; m >>= 1) v += __shfl_xor(v, m, 32);
  return v;
}

__device__ inline unsigned pack2(float a, float b) {
  union { bf16 h[2]; unsigned u; } u;
  u.h[0] = (bf16)a; u.h[1] = (bf16)b;
  return u.u;
}

// CDNA5 async global->LDS copy (16B per lane), tracked by ASYNCcnt.
// Generic pointers to __shared__ carry the wave-relative LDS offset in their
// low 32 bits (flat LDS aperture), which is what VDST expects.
__device__ inline void async_b128(void* lds, const void* g) {
  unsigned l = (unsigned)(size_t)lds;
  unsigned long long a = (unsigned long long)(size_t)g;
  asm volatile("global_load_async_to_lds_b128 %0, %1, off"
               :: "v"(l), "v"(a) : "memory");
}

// ---------------------------------------------------------------------------
// fp32 -> bf16 convert (x input)
// ---------------------------------------------------------------------------
__global__ __launch_bounds__(256) void cvt_kernel(const float* __restrict__ in,
                                                  bf16* __restrict__ out, int n4) {
  int i = blockIdx.x * blockDim.x + threadIdx.x;
  if (i >= n4) return;
  float4 f = ((const float4*)in)[i];
  uint2 p;
  p.x = pack2(f.x, f.y);
  p.y = pack2(f.z, f.w);
  ((uint2*)out)[i] = p;
}

// ---------------------------------------------------------------------------
// Weight convert + transpose: Wt[n][k] = (bf16) W[k][n]   (K=N=1024)
// ---------------------------------------------------------------------------
__global__ __launch_bounds__(256) void wconv_kernel(const float* __restrict__ W,
                                                    bf16* __restrict__ Wt,
                                                    int K, int N) {
  __shared__ float t[32][33];
  int n0 = blockIdx.x * 32, k0 = blockIdx.y * 32;
  int tx = threadIdx.x & 31, ty = threadIdx.x >> 5;
  for (int i = 0; i < 4; ++i)
    t[ty + 8 * i][tx] = W[(size_t)(k0 + ty + 8 * i) * N + n0 + tx];
  __syncthreads();
  for (int i = 0; i < 4; ++i)
    Wt[(size_t)(n0 + ty + 8 * i) * K + k0 + tx] = (bf16)t[tx][ty + 8 * i];
}

// ---------------------------------------------------------------------------
// LayerNorm (fp32 in) fused with bf16 convert. One wave per row, C=1024.
// ---------------------------------------------------------------------------
__global__ __launch_bounds__(256) void ln_kernel(const float* __restrict__ in,
                                                 const float* __restrict__ g,
                                                 const float* __restrict__ b,
                                                 bf16* __restrict__ out) {
  int row = blockIdx.x * 8 + (threadIdx.x >> 5);
  int lane = threadIdx.x & 31;
  const float* r = in + (size_t)row * C_DIM;
  float4 v[8];
  float s = 0.f;
  for (int j = 0; j < 8; ++j) {
    v[j] = *(const float4*)(r + j * 128 + lane * 4);
    s += v[j].x + v[j].y + v[j].z + v[j].w;
  }
  float mean = wsum32(s) * (1.0f / C_DIM);
  float var = 0.f;
  for (int j = 0; j < 8; ++j) {
    float4 d = v[j];
    d.x -= mean; d.y -= mean; d.z -= mean; d.w -= mean;
    var += d.x * d.x + d.y * d.y + d.z * d.z + d.w * d.w;
  }
  var = wsum32(var) * (1.0f / C_DIM);
  float rstd = rsqrtf(var + 1e-5f);
  for (int j = 0; j < 8; ++j) {
    int idx = j * 128 + lane * 4;
    float4 gg = *(const float4*)(g + idx);
    float4 bb = *(const float4*)(b + idx);
    float o0 = (v[j].x - mean) * rstd * gg.x + bb.x;
    float o1 = (v[j].y - mean) * rstd * gg.y + bb.y;
    float o2 = (v[j].z - mean) * rstd * gg.z + bb.z;
    float o3 = (v[j].w - mean) * rstd * gg.w + bb.w;
    uint2 p; p.x = pack2(o0, o1); p.y = pack2(o2, o3);
    *(uint2*)(out + (size_t)row * C_DIM + idx) = p;
  }
}

// ---------------------------------------------------------------------------
// Tiled WMMA GEMM: C[M,N] = A[M,K] @ Wt^T + bias, A bf16 [M,K], Wt bf16 [N,K].
// Block tile 128x128x32, 8 waves (wave tile 64x32).
// Double-buffered LDS staging via CDNA5 async global->LDS loads (ASYNCcnt):
// issue tile k+1, s_wait_asynccnt 4 (per-thread: only the 4 newest remain ->
// tile k done, async loads complete in order), barrier, compute.
// Epilogues: 0 fp32, 1 bf16, 2 bf16-transposed (Vt[img][n][m%seg]), 3 GELU bf16
// ---------------------------------------------------------------------------
template <int EPI>
__global__ __launch_bounds__(256) void gemm_kernel(
    const bf16* __restrict__ A, const bf16* __restrict__ Wt,
    const float* __restrict__ bias, void* __restrict__ outp, int M, int N,
    int K, int seg) {
  __shared__ bf16 sA[2][128][40];  // +8 pad keeps rows 16B-aligned
  __shared__ bf16 sB[2][128][40];
  int t = threadIdx.x;
  int wid = t >> 5, lane = t & 31, half = lane >> 4, lr = lane & 15;
  int wm = wid & 1, wn = wid >> 1;
  int m0 = blockIdx.y * 128, n0 = blockIdx.x * 128;
  int lrow = t >> 1;
  int lk = (t & 1) * 16;

  const v8f vz = {0.f, 0.f, 0.f, 0.f, 0.f, 0.f, 0.f, 0.f};
  v8f acc[4][2];
  for (int mt = 0; mt < 4; ++mt)
    for (int nt = 0; nt < 2; ++nt) acc[mt][nt] = vz;

  const bf16* gA = A + (size_t)(m0 + lrow) * K + lk;
  const bf16* gB = Wt + (size_t)(n0 + lrow) * K + lk;

  auto issue = [&](int kb, int buf) {
    async_b128(&sA[buf][lrow][lk], gA + kb);
    async_b128(&sA[buf][lrow][lk + 8], gA + kb + 8);
    async_b128(&sB[buf][lrow][lk], gB + kb);
    async_b128(&sB[buf][lrow][lk + 8], gB + kb + 8);
  };

  int nk = K >> 5;
  issue(0, 0);
  for (int ib = 0; ib < nk; ++ib) {
    int buf = ib & 1;
    if (ib + 1 < nk) {
      issue((ib + 1) << 5, buf ^ 1);
      if (ib + 2 < nk) {
        __builtin_prefetch(gA + ((ib + 2) << 5), 0, 0);
        __builtin_prefetch(gB + ((ib + 2) << 5), 0, 0);
      }
      asm volatile("s_wait_asynccnt 0x4" ::: "memory");
    } else {
      asm volatile("s_wait_asynccnt 0x0" ::: "memory");
    }
    __syncthreads();
    v16bf bfr[2];
    for (int nt = 0; nt < 2; ++nt) {
      int rr = wn * 32 + nt * 16 + lr;
      bfr[nt] = cat8(*(const v8bf*)&sB[buf][rr][16 * half],
                     *(const v8bf*)&sB[buf][rr][16 * half + 8]);
    }
    for (int mt = 0; mt < 4; ++mt) {
      int rr = wm * 64 + mt * 16 + lr;
      v16bf af = cat8(*(const v8bf*)&sA[buf][rr][16 * half],
                      *(const v8bf*)&sA[buf][rr][16 * half + 8]);
      acc[mt][0] = wmma_bf16(af, bfr[0], acc[mt][0]);
      acc[mt][1] = wmma_bf16(af, bfr[1], acc[mt][1]);
    }
    __syncthreads();
  }

  for (int nt = 0; nt < 2; ++nt) {
    int gn = n0 + wn * 32 + nt * 16 + lr;
    float bia = bias[gn];
    for (int mt = 0; mt < 4; ++mt) {
      for (int r8 = 0; r8 < 8; ++r8) {
        int gm = m0 + wm * 64 + mt * 16 + r8 + 8 * half;
        float v = acc[mt][nt][r8] + bia;
        if (EPI == 0) {
          ((float*)outp)[(size_t)gm * N + gn] = v;
        } else if (EPI == 1) {
          ((bf16*)outp)[(size_t)gm * N + gn] = (bf16)v;
        } else if (EPI == 2) {
          int img = gm / seg, kk = gm - img * seg;
          ((bf16*)outp)[(size_t)img * N * seg + (size_t)gn * seg + kk] =
              (bf16)v;
        } else {
          float ge = 0.5f * v * (1.0f + erff(v * 0.70710678118654752f));
          ((bf16*)outp)[(size_t)gm * N + gn] = (bf16)ge;
        }
      }
    }
  }
}

// ---------------------------------------------------------------------------
// Flash attention: 1 wave per (image, head, 16-q-row tile); 4 waves per block.
// Q[I*Lq, C] bf16, K[I*Lk, C] bf16, Vt[I][C][Lk] bf16 -> O[I*Lq, C] bf16.
// ---------------------------------------------------------------------------
__global__ __launch_bounds__(128) void attn_kernel(
    const bf16* __restrict__ Q, const bf16* __restrict__ Km,
    const bf16* __restrict__ Vt, bf16* __restrict__ O, int Lq, int Lk,
    float scale) {
  __shared__ bf16 P[4][16][40];
  int wid = threadIdx.x >> 5, lane = threadIdx.x & 31;
  int half = lane >> 4, lr = lane & 15;
  int qtiles = Lq >> 4;
  int g = blockIdx.x * 4 + wid;
  int img = g / (qtiles * NUM_HEAD);
  int rem = g - img * (qtiles * NUM_HEAD);
  int head = rem / qtiles;
  int qt = rem - head * qtiles;
  int q0 = img * Lq + qt * 16;

  // Q fragments (A layout, K dim = channel, 4 slices of 32)
  v16bf qf[4];
  const bf16* qp = Q + (size_t)(q0 + lr) * C_DIM + head * DH;
  for (int kc = 0; kc < 4; ++kc)
    qf[kc] = cat8(*(const v8bf*)(qp + kc * 32 + 16 * half),
                  *(const v8bf*)(qp + kc * 32 + 16 * half + 8));

  const v8f vz = {0.f, 0.f, 0.f, 0.f, 0.f, 0.f, 0.f, 0.f};
  v8f accO[8];
  float rm[8], ls[8];
  for (int d = 0; d < 8; ++d) accO[d] = vz;
  for (int r = 0; r < 8; ++r) { rm[r] = -3.0e38f; ls[r] = 0.f; }

  const bf16* kbase = Km + (size_t)img * Lk * C_DIM + head * DH;
  const bf16* vbase = Vt + ((size_t)img * C_DIM + head * DH) * (size_t)Lk;

  for (int kb = 0; kb < Lk; kb += 32) {
    // Batch all K-fragment loads first (one clause, one wait), then WMMA.
    v8bf kraw[2][8];
    for (int sub = 0; sub < 2; ++sub) {
      const bf16* kp = kbase + (size_t)(kb + sub * 16 + lr) * C_DIM + 16 * half;
      for (int kc = 0; kc < 4; ++kc) {
        kraw[sub][2 * kc] = *(const v8bf*)(kp + kc * 32);
        kraw[sub][2 * kc + 1] = *(const v8bf*)(kp + kc * 32 + 8);
      }
    }
    v8f s[2];
    for (int sub = 0; sub < 2; ++sub) {
      v8f sa = vz;
      for (int kc = 0; kc < 4; ++kc)
        sa = wmma_bf16(qf[kc], cat8(kraw[sub][2 * kc], kraw[sub][2 * kc + 1]),
                       sa);
      s[sub] = sa * scale;
    }
    // Issue V-tile loads now so their latency hides behind the softmax math.
    v8bf vraw[16];
    for (int dt = 0; dt < 8; ++dt) {
      const bf16* vp = vbase + (size_t)(dt * 16 + lr) * Lk + kb + 16 * half;
      vraw[2 * dt] = *(const v8bf*)vp;
      vraw[2 * dt + 1] = *(const v8bf*)(vp + 8);
    }
    // online softmax per row (rows live in 16-lane halves of C layout)
    float corr[8];
    for (int r = 0; r < 8; ++r) {
      float tm = hmax16(fmaxf(s[0][r], s[1][r]));
      float nm = fmaxf(rm[r], tm);
      corr[r] = expf(rm[r] - nm);
      float p0 = expf(s[0][r] - nm);
      float p1 = expf(s[1][r] - nm);
      s[0][r] = p0; s[1][r] = p1;
      float ps = hsum16(p0 + p1);
      ls[r] = ls[r] * corr[r] + ps;
      rm[r] = nm;
    }
    for (int d = 0; d < 8; ++d)
      for (int r = 0; r < 8; ++r) accO[d][r] *= corr[r];
    // bounce P through LDS: C layout -> A layout (LDS is in-order per wave)
    for (int r = 0; r < 8; ++r) {
      P[wid][r + 8 * half][lr] = (bf16)s[0][r];
      P[wid][r + 8 * half][16 + lr] = (bf16)s[1][r];
    }
    asm volatile("s_wait_dscnt 0x0" ::: "memory");
    v16bf pa = cat8(*(const v8bf*)&P[wid][lr][16 * half],
                    *(const v8bf*)&P[wid][lr][16 * half + 8]);
    for (int dt = 0; dt < 8; ++dt)
      accO[dt] =
          wmma_bf16(pa, cat8(vraw[2 * dt], vraw[2 * dt + 1]), accO[dt]);
  }
  float inv[8];
  for (int r = 0; r < 8; ++r) inv[r] = 1.0f / ls[r];
  for (int dt = 0; dt < 8; ++dt)
    for (int r = 0; r < 8; ++r)
      O[(size_t)(q0 + r + 8 * half) * C_DIM + head * DH + dt * 16 + lr] =
          (bf16)(accO[dt][r] * inv[r]);
}

// ---------------------------------------------------------------------------
// Residual updates
// ---------------------------------------------------------------------------
__global__ __launch_bounds__(256) void resid2_kernel(
    float* __restrict__ x, const float* __restrict__ h1,
    const float* __restrict__ h2, const float* __restrict__ s1,
    const float* __restrict__ s2, int n4) {
  int i = blockIdx.x * blockDim.x + threadIdx.x;
  if (i >= n4) return;
  int c = (i * 4) & (C_DIM - 1);
  float4 xv = ((float4*)x)[i];
  float4 a = ((const float4*)h1)[i];
  float4 b = ((const float4*)h2)[i];
  float4 sa = *(const float4*)(s1 + c);
  float4 sb = *(const float4*)(s2 + c);
  xv.x += sa.x * a.x + sb.x * b.x;
  xv.y += sa.y * a.y + sb.y * b.y;
  xv.z += sa.z * a.z + sb.z * b.z;
  xv.w += sa.w * a.w + sb.w * b.w;
  ((float4*)x)[i] = xv;
}

__global__ __launch_bounds__(256) void residout_kernel(
    const float* __restrict__ x, const float* __restrict__ h3,
    const float* __restrict__ s3, float* __restrict__ out, int n4) {
  int i = blockIdx.x * blockDim.x + threadIdx.x;
  if (i >= n4) return;
  int c = (i * 4) & (C_DIM - 1);
  float4 xv = ((const float4*)x)[i];
  float4 a = ((const float4*)h3)[i];
  float4 sa = *(const float4*)(s3 + c);
  xv.x += sa.x * a.x;
  xv.y += sa.y * a.y;
  xv.z += sa.z * a.z;
  xv.w += sa.w * a.w;
  ((float4*)out)[i] = xv;
}

// ---------------------------------------------------------------------------
// Host launcher
// Input order (flattened setup_inputs dict, insertion order):
//  0 x[16384,1024] 1 gf[32,256,1024]
//  2/3 emb W,b  4/5 sca_q  6/7 sca_k  8/9 sca_v  10/11 sca_u
//  12/13 gca_q 14/15 gca_k 16/17 gca_v 18/19 gca_u 20/21 fc1 22/23 fc2
//  24/25 norm1 26/27 norm1_1 28/29 norm2 30/31 norm3
//  32 scale1 33 scale2 34 scale3 35 num_inst_per_image
// ---------------------------------------------------------------------------
extern "C" void kernel_launch(void* const* d_in, const int* in_sizes, int n_in,
                              void* d_out, int out_size, void* d_ws,
                              size_t ws_size, hipStream_t stream) {
  (void)in_sizes; (void)n_in; (void)out_size; (void)ws_size;
  const int B = 16384;    // I*N
  const int MG = 8192;    // I*G
  const int I = 32, Nn = 512, G = 256;
  const float scale = 1.0f / 32.0f;  // 1/sqrt(1024)

  const float* xin = (const float*)d_in[0];
  const float* gf  = (const float*)d_in[1];

  char* ws = (char*)d_ws;
  size_t off = 0;
  auto alloc = [&](size_t bytes) {
    char* p = ws + off;
    off += (bytes + 255) & ~(size_t)255;
    return p;
  };
  bf16* Wt[11];
  const int widx[11] = {2, 4, 6, 8, 10, 12, 14, 16, 18, 20, 22};
  for (int i = 0; i < 11; ++i)
    Wt[i] = (bf16*)alloc((size_t)C_DIM * C_DIM * sizeof(bf16));
  float* X  = (float*)alloc((size_t)B * C_DIM * sizeof(float));
  bf16* Ab  = (bf16*)alloc((size_t)B * C_DIM * sizeof(bf16));
  bf16* Qb  = (bf16*)alloc((size_t)B * C_DIM * sizeof(bf16));
  bf16* Kb  = (bf16*)alloc((size_t)B * C_DIM * sizeof(bf16));
  bf16* Vb  = (bf16*)alloc((size_t)B * C_DIM * sizeof(bf16));
  float* F1 = (float*)alloc((size_t)B * C_DIM * sizeof(float));
  float* F2 = (float*)alloc((size_t)B * C_DIM * sizeof(float));
  bf16* Gb  = (bf16*)F2;                    // h2 (LN of gf), dead before F2 use
  bf16* KGb = Vb;                           // kg [8192,1024]
  bf16* VGb = Vb + (size_t)MG * C_DIM;      // vtg [I][C][256]

  const dim3 gB(C_DIM / 128, B / 128);   // (8,128)
  const dim3 gG(C_DIM / 128, MG / 128);  // (8,64)
  const int n4 = B * C_DIM / 4;

  // weight prep
  for (int i = 0; i < 11; ++i)
    wconv_kernel<<<dim3(32, 32), 256, 0, stream>>>((const float*)d_in[widx[i]],
                                                   Wt[i], C_DIM, C_DIM);
  // embedding
  cvt_kernel<<<(n4 + 255) / 256, 256, 0, stream>>>(xin, Ab, n4);
  gemm_kernel<0><<<gB, 256, 0, stream>>>(Ab, Wt[0], (const float*)d_in[3], X, B,
                                         C_DIM, C_DIM, 0);
  // self attention branch
  ln_kernel<<<B / 8, 256, 0, stream>>>(X, (const float*)d_in[24],
                                       (const float*)d_in[25], Ab);
  gemm_kernel<1><<<gB, 256, 0, stream>>>(Ab, Wt[1], (const float*)d_in[5], Qb,
                                         B, C_DIM, C_DIM, 0);
  gemm_kernel<1><<<gB, 256, 0, stream>>>(Ab, Wt[2], (const float*)d_in[7], Kb,
                                         B, C_DIM, C_DIM, 0);
  gemm_kernel<2><<<gB, 256, 0, stream>>>(Ab, Wt[3], (const float*)d_in[9], Vb,
                                         B, C_DIM, C_DIM, Nn);
  attn_kernel<<<(I * NUM_HEAD * (Nn / 16)) / 4, 128, 0, stream>>>(
      Qb, Kb, Vb, Ab, Nn, Nn, scale);
  gemm_kernel<0><<<gB, 256, 0, stream>>>(Ab, Wt[4], (const float*)d_in[11], F1,
                                         B, C_DIM, C_DIM, 0);
  // cross attention branch
  ln_kernel<<<B / 8, 256, 0, stream>>>(X, (const float*)d_in[26],
                                       (const float*)d_in[27], Ab);
  ln_kernel<<<MG / 8, 256, 0, stream>>>(gf, (const float*)d_in[28],
                                        (const float*)d_in[29], Gb);
  gemm_kernel<1><<<gB, 256, 0, stream>>>(Ab, Wt[5], (const float*)d_in[13], Qb,
                                         B, C_DIM, C_DIM, 0);
  gemm_kernel<1><<<gG, 256, 0, stream>>>(Gb, Wt[6], (const float*)d_in[15], KGb,
                                         MG, C_DIM, C_DIM, 0);
  gemm_kernel<2><<<gG, 256, 0, stream>>>(Gb, Wt[7], (const float*)d_in[17], VGb,
                                         MG, C_DIM, C_DIM, G);
  attn_kernel<<<(I * NUM_HEAD * (Nn / 16)) / 4, 128, 0, stream>>>(
      Qb, KGb, VGb, Kb, Nn, G, scale);
  gemm_kernel<0><<<gB, 256, 0, stream>>>(Kb, Wt[8], (const float*)d_in[19], F2,
                                         B, C_DIM, C_DIM, 0);
  // residual 1
  resid2_kernel<<<(n4 + 255) / 256, 256, 0, stream>>>(
      X, F1, F2, (const float*)d_in[32], (const float*)d_in[33], n4);
  // MLP
  ln_kernel<<<B / 8, 256, 0, stream>>>(X, (const float*)d_in[30],
                                       (const float*)d_in[31], Ab);
  gemm_kernel<3><<<gB, 256, 0, stream>>>(Ab, Wt[9], (const float*)d_in[21], Qb,
                                         B, C_DIM, C_DIM, 0);
  gemm_kernel<0><<<gB, 256, 0, stream>>>(Qb, Wt[10], (const float*)d_in[23], F1,
                                         B, C_DIM, C_DIM, 0);
  residout_kernel<<<(n4 + 255) / 256, 256, 0, stream>>>(
      X, F1, (const float*)d_in[34], (float*)d_out, n4);
}
